// PredNet_29549374996837
// MI455X (gfx1250) — matrix operand order repaired and checked by reading, not compiled, not run on hardware
//
#include <hip/hip_runtime.h>

// ---------------------------------------------------------------------------
// MI455X (gfx1250) implementation.
//
// Roofline: ~4.5 GFLOP total vs 184 MB of mandatory input traffic ->
// HBM-bound (~8 us at 23.3 TB/s). Strategy: one workgroup per sample image,
// whole resblock pipeline resident in LDS, convolutions as implicit-GEMM on
// v_wmma_f32_16x16x32_f16 (wave32 WMMA), BN folded into f16-packed weights
// that stay L2-resident across all 4096 workgroups. Bulk input traffic is
// moved by the Tensor Data Mover (tensor_load_to_lds, TENSORcnt) so the
// HBM stream never round-trips through VGPRs.
//
// d_in flattening assumption (insertion order of the setup_inputs dict):
//   0: past_traj (512*8*1*11*32*32)   1: obs (512*32*32*6)
//   char block i (i=0..4) at base 2+9i: w1,g1,b1,w2,g2,b2,ws,gs,bs
//   47: lstm.Wih  48: lstm.Whh (unused, h0=0)  49: bih  50: bhh
//   51: fc1.w  52: fc1.b
//   pred block j (j=0..1) at base 53+9j: w1,g1,b1,w2,g2,b2,ws,gs,bs
//   71: fc.w  72: fc.b
// d_out: out (512*5) then e_char (512*2), f32.
// ---------------------------------------------------------------------------

typedef _Float16 v16h __attribute__((ext_vector_type(16)));
typedef _Float16 v8h  __attribute__((ext_vector_type(8)));
typedef float    v8f  __attribute__((ext_vector_type(8)));
typedef unsigned int u32x4 __attribute__((ext_vector_type(4)));
typedef int          i32x8 __attribute__((ext_vector_type(8)));
typedef int          i32x4 __attribute__((ext_vector_type(4)));

#define NW  4     // waves per workgroup (blockDim = 128, wave32)
#define KCH 128   // K-chunk staged in LDS per tile (multiple of 32)

struct ConvTab {
  const _Float16* W[21];  // packed f16 weights [Npad][Kpad], BN gamma folded in
  const float*    B[21];  // BN beta (bias), [Npad]
};

__device__ __forceinline__ float sigmoidf_(float x) { return 1.f / (1.f + __expf(-x)); }

// ---------------------------------------------------------------------------
// Tensor Data Mover: DMA `nelem` contiguous 4-byte elements global -> LDS.
// D# per ISA ch.8: group0 {count=1, lds_addr, global_addr, type=2}; group1
// {data_size=4B, 1-D: tensor_dim0 = tile_dim0 = nelem, tile_dim1 = 1,
//  tensor_dim0_stride = nelem}; groups 2/3 zero (<=2-D tensor, VADDR2/3).
// Wave-level op (EXEC ignored) -> issue from one thread, wait TENSORcnt==0.
// Generic LDS pointer: low 32 bits are the LDS byte offset (aperture rule).
// This toolchain exposes the 6-arg builtin (extra zero-filled group).
// ---------------------------------------------------------------------------
__device__ __forceinline__ void tdm_load_1d(void* lds_dst, const void* gsrc,
                                            unsigned nelem) {
  const unsigned lds_addr = (unsigned)(size_t)lds_dst;
  const unsigned long long ga = (unsigned long long)(size_t)gsrc;
  u32x4 g0;
  g0.x = 1u;                                    // count=1 valid descriptor
  g0.y = lds_addr;                              // lds_addr      [63:32]
  g0.z = (unsigned)(ga & 0xffffffffu);          // global_addr   [95:64]
  g0.w = (unsigned)((ga >> 32) & 0x01ffffffu)   // global_addr   [120:96]
       | (2u << 30);                            // type=2        [127:126]
  i32x8 g1;
  g1[0] = (2 << 16);                            // data_size=2 -> 4B; wg_mask=0
  g1[1] = (int)((nelem & 0xffffu) << 16);       // tensor_dim0 lo16  [63:48]
  g1[2] = (int)((nelem >> 16) & 0xffffu)        // tensor_dim0 hi16  [79:64]
        | (1 << 16);                            // tensor_dim1=1 lo  [95:80]
  g1[3] = (int)((nelem & 0xffffu) << 16);       // tile_dim0        [127:112]
  g1[4] = 1;                                    // tile_dim1=1      [143:128]
  g1[5] = (int)nelem;                           // dim0_stride lo32 [191:160]
  g1[6] = 0;
  g1[7] = 0;
  const i32x4 z4 = {0, 0, 0, 0};
  const i32x8 z8 = {0, 0, 0, 0, 0, 0, 0, 0};
  __builtin_amdgcn_tensor_load_to_lds(g0, g1, z4, z4, z8, 0);
}

// ---------------------------------------------------------------------------
// Pack kernel: wp[n][k] = w[n][k] * g[n]  (f32 -> f16, zero-padded), bp = b.
// ---------------------------------------------------------------------------
__global__ void pack_conv(const float* __restrict__ w, const float* __restrict__ g,
                          const float* __restrict__ b,
                          _Float16* __restrict__ wp, float* __restrict__ bp,
                          int cout, int K, int Kpad, int Npad) {
  const int i = blockIdx.x * blockDim.x + threadIdx.x;
  const int total = Npad * Kpad;
  if (i < total) {
    const int n = i / Kpad, k = i - n * Kpad;
    float v = 0.f;
    if (n < cout && k < K) v = w[n * K + k] * g[n];
    wp[i] = (_Float16)v;
  }
  if (i < Npad) bp[i] = (i < cout) ? b[i] : 0.f;
}

// ---------------------------------------------------------------------------
// Implicit-GEMM conv on WMMA. One wave per 16(M=out pixels) x 16(N=cout) tile.
// A (16xK im2col patch) staged per-wave in LDS following the gfx1250 16-bit A
// layout; B read straight from packed global [Npad][Kpad] (lane r = column,
// K contiguous, 16-byte loads). Output -> LDS f32 [cout][Hout*Wout] + bias.
// ---------------------------------------------------------------------------
__device__ void conv_wmma(const _Float16* xin, int cin, int Hin, int Win,
                          const _Float16* Wg, const float* bias,
                          int cout, int ksz, int stride, int pad,
                          int Hout, int Wout,
                          _Float16* patch_all, float* yout) {
  const int lane = threadIdx.x & 31;
  const int wave = threadIdx.x >> 5;
  _Float16* patch = patch_all + wave * (16 * KCH);
  const int M    = Hout * Wout;
  const int K    = cin * ksz * ksz;
  const int Kpad = (K + 31) & ~31;
  const int Mt   = (M + 15) >> 4;
  const int Nt   = (cout + 15) >> 4;
  const int half_id = lane >> 4;
  const int r       = lane & 15;

  __builtin_prefetch(Wg, 0, 3);  // global_prefetch_b8 on the weight tile

  for (int tile = wave; tile < Mt * Nt; tile += NW) {   // lane-uniform: EXEC all-1s at WMMA
    const int tm = tile % Mt;
    const int tn = tile / Mt;
    v8f acc = {};
    for (int k0 = 0; k0 < Kpad; k0 += KCH) {
      const int kc = (Kpad - k0) < KCH ? (Kpad - k0) : KCH;  // multiple of 32
      __builtin_amdgcn_wave_barrier();  // WAR fence vs previous chunk's frag loads
      // Build im2col patch rows (zero-fill padding / out-of-range).
      for (int idx = lane; idx < 16 * kc; idx += 32) {
        const int mm = idx / kc;
        const int kk = idx - mm * kc;
        const int m  = (tm << 4) + mm;
        const int k  = k0 + kk;
        _Float16 v = (_Float16)0.f;
        if (m < M && k < K) {
          const int oh = m / Wout, ow = m - oh * Wout;
          int c, ih, iw;
          if (ksz == 3) {
            c = k / 9;
            const int rr = k - c * 9;
            const int kh = rr / 3, kw = rr - kh * 3;
            ih = oh * stride + kh - pad;
            iw = ow * stride + kw - pad;
          } else {
            c = k; ih = oh * stride; iw = ow * stride;
          }
          if ((unsigned)ih < (unsigned)Hin && (unsigned)iw < (unsigned)Win)
            v = xin[(c * Hin + ih) * Win + iw];
        }
        patch[mm * KCH + kk] = v;
      }
      __builtin_amdgcn_wave_barrier();  // RAW fence: cross-lane LDS visibility (in-order DS)
      const _Float16* Arow = patch + r * KCH;
      const _Float16* Brow = Wg + (size_t)((tn << 4) + r) * Kpad + k0;
      for (int kk = 0; kk < kc; kk += 32) {
        v8h lo = *(const v8h*)(Arow + kk + half_id * 8);        // K = hk..hk+7
        v8h hi = *(const v8h*)(Arow + kk + 16 + half_id * 8);   // K = 16+hk..+7
        v16h a, b;
#pragma unroll
        for (int i = 0; i < 8; ++i) { a[i] = lo[i]; a[i + 8] = hi[i]; }
        v8h blo = *(const v8h*)(Brow + kk + half_id * 8);       // col r, K = hk..hk+7
        v8h bhi = *(const v8h*)(Brow + kk + 16 + half_id * 8);  // col r, K = 16+hk..+7
#pragma unroll
        for (int i = 0; i < 8; ++i) { b[i] = blo[i]; b[i + 8] = bhi[i]; }
        acc = __builtin_amdgcn_wmma_f32_16x16x32_f16(
            /*neg_a=*/false, a, /*neg_b=*/false, b,
            /*c_mod=*/(short)0, acc, /*reuse_a=*/false, /*reuse_b=*/false);
      }
    }
    // D layout: lane(half_id,r), VGPR v -> (M = v + 8*half_id, N = r)
    const int n = (tn << 4) + r;
    if (n < cout) {
      const float bb = bias[n];
#pragma unroll
      for (int v2 = 0; v2 < 8; ++v2) {
        const int m = (tm << 4) + v2 + (half_id << 3);
        if (m < M) yout[n * M + m] = acc[v2] + bb;
      }
    }
  }
}

// ---------------------------------------------------------------------------
// Resblock: relu(bn2(conv2(relu(bn1(conv1 x)))) + bn_s(conv1x1 x)).
// BN folded into weights (scale) + bias arrays. All buffers in LDS.
// ---------------------------------------------------------------------------
__device__ void resblock(const _Float16* xin, int cin, int Hin, int Win, int stride, int cout,
                         const ConvTab& T, int l0,
                         _Float16* y1, _Float16* xout, float* t1, float* t2,
                         _Float16* patch) {
  const int Ho = (Hin + 2 - 3) / stride + 1;   // matches 1x1 stride-s shortcut dims here
  const int Wo = (Win + 2 - 3) / stride + 1;
  const int n  = cout * Ho * Wo;

  conv_wmma(xin, cin, Hin, Win, T.W[l0], T.B[l0], cout, 3, stride, 1, Ho, Wo, patch, t1);
  __syncthreads();
  for (int i = threadIdx.x; i < n; i += blockDim.x) {
    const float v = t1[i];
    y1[i] = (_Float16)(v > 0.f ? v : 0.f);
  }
  __syncthreads();
  conv_wmma(xin, cin, Hin, Win, T.W[l0 + 2], T.B[l0 + 2], cout, 1, stride, 0, Ho, Wo, patch, t2);
  __syncthreads();
  conv_wmma(y1, cout, Ho, Wo, T.W[l0 + 1], T.B[l0 + 1], cout, 3, 1, 1, Ho, Wo, patch, t1);
  __syncthreads();
  for (int i = threadIdx.x; i < n; i += blockDim.x) {
    const float v = t1[i] + t2[i];
    xout[i] = (_Float16)(v > 0.f ? v : 0.f);
  }
  __syncthreads();
}

// ---------------------------------------------------------------------------
// CharNet: one workgroup per (b,p) image. TDM image load -> 5 resblocks ->
// LSTM cell -> pool -> fc1 -> e_all[bp][2]. ~52 KB LDS.
// ---------------------------------------------------------------------------
__global__ __launch_bounds__(128) void charnet_kernel(
    const float* __restrict__ past, ConvTab T,
    const float* __restrict__ Wih, const float* __restrict__ bih,
    const float* __restrict__ bhh,
    const float* __restrict__ fc1w, const float* __restrict__ fc1b,
    float* __restrict__ e_all) {
  __shared__ _Float16 s_a0[11 * 32 * 32];
  __shared__ _Float16 s_a1[1024];
  __shared__ _Float16 s_y1[1024];
  __shared__ float    s_t1[1024];
  __shared__ float    s_t2[1024];
  __shared__ alignas(16) _Float16 s_patch[NW * 16 * KCH];  // also TDM staging (16 KB)
  __shared__ float    s_g[256];
  __shared__ float    s_h[64];
  __shared__ float    s_p[8];

  const int bp = blockIdx.x;
  const float* img = past + (size_t)bp * (11 * 32 * 32);

  // TDM: stream the 45 KB f32 image into LDS in 4 chunks through s_patch
  // (free before the first conv), then convert f32 -> f16 activations.
  const float* stage = (const float*)s_patch;
  for (int chunk = 0; chunk < 4; ++chunk) {
    if (threadIdx.x == 0) {
      tdm_load_1d(s_patch, img + chunk * 2816, 2816u);
      __builtin_amdgcn_s_wait_tensorcnt(0);
    }
    __syncthreads();
    for (int i = threadIdx.x; i < 2816; i += blockDim.x)
      s_a0[chunk * 2816 + i] = (_Float16)stage[i];
    __syncthreads();
  }

  resblock(s_a0, 11, 32, 32, 2,  4, T,  0, s_y1, s_a1, s_t1, s_t2, s_patch);
  resblock(s_a1,  4, 16, 16, 2,  8, T,  3, s_y1, s_a0, s_t1, s_t2, s_patch);
  resblock(s_a0,  8,  8,  8, 2, 16, T,  6, s_y1, s_a1, s_t1, s_t2, s_patch);
  resblock(s_a1, 16,  4,  4, 2, 32, T,  9, s_y1, s_a0, s_t1, s_t2, s_patch);
  resblock(s_a0, 32,  2,  2, 2, 32, T, 12, s_y1, s_a1, s_t1, s_t2, s_patch);
  // features (already ReLU'd): s_a1[0..31]

  // LSTM cell with h0=c0=0: gates = Wih@x + bih + bhh; forget gate unused.
  for (int g = threadIdx.x; g < 256; g += blockDim.x) {
    float s = bih[g] + bhh[g];
#pragma unroll
    for (int k = 0; k < 32; ++k) s += Wih[g * 32 + k] * (float)s_a1[k];
    s_g[g] = s;
  }
  __syncthreads();
  for (int h = threadIdx.x; h < 64; h += blockDim.x) {
    const float c = sigmoidf_(s_g[h]) * tanhf(s_g[128 + h]);  // i * tanh(g)
    s_h[h] = sigmoidf_(s_g[192 + h]) * tanhf(c);              // o * tanh(c)
  }
  __syncthreads();
  if (threadIdx.x < 8) {  // AvgPool1d(8)
    float s = 0.f;
#pragma unroll
    for (int j = 0; j < 8; ++j) s += s_h[threadIdx.x * 8 + j];
    s_p[threadIdx.x] = s * 0.125f;
  }
  __syncthreads();
  if (threadIdx.x < 2) {  // fc1
    float s = fc1b[threadIdx.x];
#pragma unroll
    for (int j = 0; j < 8; ++j) s += fc1w[threadIdx.x * 8 + j] * s_p[j];
    e_all[bp * 2 + threadIdx.x] = s;
  }
}

// ---------------------------------------------------------------------------
// PredNet head: one workgroup per sample. ~56 KB LDS.
// ---------------------------------------------------------------------------
__global__ __launch_bounds__(128) void prednet_kernel(
    const float* __restrict__ obs, ConvTab T,
    const float* __restrict__ e_all,
    const float* __restrict__ fcw, const float* __restrict__ fcb,
    float* __restrict__ out) {
  __shared__ _Float16 s_xc[8 * 32 * 32];
  __shared__ _Float16 s_a1[2048];
  __shared__ _Float16 s_y1[2048];
  __shared__ float    s_t1[2048];
  __shared__ float    s_t2[2048];
  __shared__ alignas(16) _Float16 s_patch[NW * 16 * KCH];  // also TDM staging (16 KB)
  __shared__ float    s_e[2];
  __shared__ float    s_pool[16];
  __shared__ float    s_log[5];

  const int b = blockIdx.x;
  if (threadIdx.x < 2) {  // e_char = sum over P; also second tuple output
    float s = 0.f;
    for (int p = 0; p < 8; ++p) s += e_all[(b * 8 + p) * 2 + threadIdx.x];
    s_e[threadIdx.x] = s;
    out[512 * 5 + b * 2 + threadIdx.x] = s;
  }
  __syncthreads();

  // TDM: obs sample (24 KB f32, HWC) in 2 chunks through s_patch, scatter
  // into NCHW f16 channels 2..7 of the concat input.
  const float* ob = obs + (size_t)b * (32 * 32 * 6);
  const float* stage = (const float*)s_patch;
  for (int chunk = 0; chunk < 2; ++chunk) {
    if (threadIdx.x == 0) {
      tdm_load_1d(s_patch, ob + chunk * 3072, 3072u);
      __builtin_amdgcn_s_wait_tensorcnt(0);
    }
    __syncthreads();
    for (int i = threadIdx.x; i < 3072; i += blockDim.x) {
      const int j  = chunk * 3072 + i;       // flat (h,w,c)
      const int hw = j / 6, ch = j - hw * 6;
      s_xc[(2 + ch) * 1024 + hw] = (_Float16)stage[i];
    }
    __syncthreads();
  }
  for (int i = threadIdx.x; i < 2048; i += blockDim.x)  // channels 0,1 = e_char
    s_xc[i] = (_Float16)s_e[i >> 10];
  __syncthreads();

  resblock(s_xc, 8, 32, 32, 2,  8, T, 15, s_y1, s_a1, s_t1, s_t2, s_patch);
  resblock(s_a1, 8, 16, 16, 2, 16, T, 18, s_y1, s_xc, s_t1, s_t2, s_patch);
  // s_xc now holds (16, 8, 8)

  if (threadIdx.x < 16) {  // crop [0:6,0:6], mean
    float s = 0.f;
    for (int h = 0; h < 6; ++h)
      for (int w = 0; w < 6; ++w)
        s += (float)s_xc[threadIdx.x * 64 + h * 8 + w];
    s_pool[threadIdx.x] = s * (1.f / 36.f);
  }
  __syncthreads();
  if (threadIdx.x < 5) {
    float s = fcb[threadIdx.x];
#pragma unroll
    for (int c = 0; c < 16; ++c) s += fcw[threadIdx.x * 16 + c] * s_pool[c];
    s_log[threadIdx.x] = s;
  }
  __syncthreads();
  if (threadIdx.x == 0) {  // 5-way softmax
    float m = s_log[0];
    for (int o = 1; o < 5; ++o) m = fmaxf(m, s_log[o]);
    float z = 0.f, e[5];
    for (int o = 0; o < 5; ++o) { e[o] = __expf(s_log[o] - m); z += e[o]; }
    const float inv = 1.f / z;
    for (int o = 0; o < 5; ++o) out[b * 5 + o] = e[o] * inv;
  }
}

// ---------------------------------------------------------------------------
extern "C" void kernel_launch(void* const* d_in, const int* in_sizes, int n_in,
                              void* d_out, int out_size, void* d_ws, size_t ws_size,
                              hipStream_t stream) {
  (void)in_sizes; (void)n_in; (void)out_size; (void)ws_size;
  // 7 "blocks": 5 char + 2 pred; each has conv1(3x3), conv2(3x3), ws(1x1).
  const int cins[7]  = {11, 4, 8, 16, 32, 8, 8};
  const int couts[7] = { 4, 8, 16, 32, 32, 8, 16};
  const int bases[7] = { 2, 11, 20, 29, 38, 53, 62};

  int wi_[21], gi_[21], bi_[21], co_[21], K_[21], Kp_[21], Np_[21];
  size_t wo_[21];
  size_t off = 0;
  int l = 0;
  for (int blk = 0; blk < 7; ++blk) {
    const int ci = cins[blk], co = couts[blk], base = bases[blk];
    const int Ks[3]  = {ci * 9, co * 9, ci};
    const int wis[3] = {base, base + 3, base + 6};
    for (int j = 0; j < 3; ++j, ++l) {
      wi_[l] = wis[j]; gi_[l] = wis[j] + 1; bi_[l] = wis[j] + 2;
      co_[l] = co;
      K_[l]  = Ks[j];
      Kp_[l] = (Ks[j] + 31) & ~31;
      Np_[l] = (co + 15) & ~15;
      wo_[l] = off;
      off += (size_t)Np_[l] * Kp_[l] * sizeof(_Float16);
    }
  }

  char* wsb = (char*)d_ws;
  size_t boff = (off + 255) & ~(size_t)255;
  ConvTab T;
  for (l = 0; l < 21; ++l) {
    _Float16* wp = (_Float16*)(wsb + wo_[l]);
    float*    bp = (float*)(wsb + boff);
    boff += (size_t)Np_[l] * sizeof(float);
    T.W[l] = wp;
    T.B[l] = bp;
    const int total = Np_[l] * Kp_[l];
    pack_conv<<<(total + 255) / 256, 256, 0, stream>>>(
        (const float*)d_in[wi_[l]], (const float*)d_in[gi_[l]],
        (const float*)d_in[bi_[l]], wp, bp, co_[l], K_[l], Kp_[l], Np_[l]);
  }
  float* e_all = (float*)(wsb + ((boff + 255) & ~(size_t)255));  // 4096*2 f32

  charnet_kernel<<<512 * 8, 128, 0, stream>>>(
      (const float*)d_in[0], T,
      (const float*)d_in[47], (const float*)d_in[49], (const float*)d_in[50],
      (const float*)d_in[51], (const float*)d_in[52], e_all);

  prednet_kernel<<<512, 128, 0, stream>>>(
      (const float*)d_in[1], T, e_all,
      (const float*)d_in[71], (const float*)d_in[72], (float*)d_out);
}
